// zzz_2_15925738733989
// MI455X (gfx1250) — matrix operand order repaired and checked by reading, not compiled, not run on hardware
//
#include <hip/hip_runtime.h>
#include <hip/hip_bf16.h>

// ---------------- problem dims ----------------
#define S_LEN    215
#define BATCH_N  512
#define D_INP_N  36
#define F_IN_N   860      // S_LEN * 4
#define D_T_N    160
#define NHEAD_N  4
#define DH_N     40
#define NHID_N   128
#define D_FIN_N  196

typedef __attribute__((ext_vector_type(16))) _Float16 v16h;
typedef __attribute__((ext_vector_type(8)))  _Float16 v8h;
typedef __attribute__((ext_vector_type(8)))  float    v8f;

// =====================================================================
// f32 -> padded f16 conversions (zero-fill padding so the WMMA K-loop
// needs no bounds checks at all).
// =====================================================================
__global__ __launch_bounds__(256) void cvt16(
    const float* __restrict__ in, _Float16* __restrict__ out,
    int R, int C, int Cp, long total)
{
  long gid = (long)blockIdx.x * 256 + threadIdx.x;
  if (gid >= total) return;
  int r = (int)(gid / Cp), c = (int)(gid % Cp);
  out[gid] = (r < R && c < C) ? (_Float16)in[(size_t)r * C + c] : (_Float16)0.f;
}

// out is Bt: (Np, Kp) row-major from B (K, N) row-major
__global__ __launch_bounds__(256) void cvt16T(
    const float* __restrict__ in, _Float16* __restrict__ out,
    int K, int N, int Kp, long total)
{
  long gid = (long)blockIdx.x * 256 + threadIdx.x;
  if (gid >= total) return;
  int n = (int)(gid / Kp), k = (int)(gid % Kp);
  out[gid] = (k < K && n < N) ? (_Float16)in[(size_t)k * N + n] : (_Float16)0.f;
}

// =====================================================================
// WMMA GEMM on pre-converted, zero-padded f16 operands:
//   C(MxN,f32) = A16(Mp x Kp) @ Bt16(Np x Kp)^T  (+bias, +relu)
// One wave per 16x16 tile; branch-free K loop; all fragment loads are
// contiguous 16B (global_load_b128).
// =====================================================================
template<bool BIAS, bool RELU>
__global__ __launch_bounds__(256) void wmma_gemm(
    const _Float16* __restrict__ A, const _Float16* __restrict__ Bt,
    const float* __restrict__ bias, float* __restrict__ C,
    int M, int N, int Kp)
{
  const int lane   = threadIdx.x & 31;
  const int wave   = threadIdx.x >> 5;
  const int tilesN = (N + 15) >> 4;
  const int tilesM = (M + 15) >> 4;
  const long tile  = (long)blockIdx.x * 8 + wave;
  if (tile >= (long)tilesM * tilesN) return;      // wave-uniform: EXEC stays full
  const int tM = (int)(tile / tilesN);
  const int tN = (int)(tile % tilesN);

  const int mrow = lane & 15;
  const int kh   = lane >> 4;                         // K-half carried by this lane
  const _Float16* arow = A  + (size_t)(tM * 16 + mrow) * Kp;
  const _Float16* brow = Bt + (size_t)(tN * 16 + mrow) * Kp;

  v8f acc = {};
#pragma unroll 2
  for (int k0 = 0; k0 < Kp; k0 += 32) {
    // A 16x32 frag: lanes 0-15 K[0..7 | 16..23], lanes 16-31 K[8..15 | 24..31]
    v8h a0 = *(const v8h*)(arow + k0 + kh * 8);
    v8h a1 = *(const v8h*)(arow + k0 + 16 + kh * 8);
    // B 32x16 frag: lanes 0-15 K[0..7 | 16..23], lanes 16-31 K[8..15 | 24..31] (contig in Bt)
    v8h b0 = *(const v8h*)(brow + k0 + kh * 8);
    v8h b1 = *(const v8h*)(brow + k0 + 16 + kh * 8);
    __builtin_prefetch((const void*)(arow + k0 + 128), 0, 1);
    __builtin_prefetch((const void*)(brow + k0 + 128), 0, 1);
    v16h af, bf;
#pragma unroll
    for (int i = 0; i < 8; ++i) {
      af[i] = a0[i]; af[i + 8] = a1[i];
      bf[i] = b0[i]; bf[i + 8] = b1[i];
    }
    acc = __builtin_amdgcn_wmma_f32_16x16x32_f16(false, af, false, bf,
                                                 (short)0, acc, false, false);
  }
#pragma unroll
  for (int r = 0; r < 8; ++r) {
    int orow = tM * 16 + r + 8 * kh;
    int ocol = tN * 16 + mrow;
    if (orow < M && ocol < N) {
      float v = acc[r];
      if (BIAS) v += bias[ocol];
      if (RELU) v = v > 0.f ? v : 0.f;
      C[(size_t)orow * N + ocol] = v;
    }
  }
}

// =====================================================================
// x[b,i,s*4+j] = relu(src[s,b,i] * R_u[i*4+j])
// =====================================================================
__global__ __launch_bounds__(256) void build_x(
    const float* __restrict__ src, const float* __restrict__ R_u,
    float* __restrict__ X)
{
  long gid = (long)blockIdx.x * 256 + threadIdx.x;
  const long TOT = (long)BATCH_N * D_INP_N * F_IN_N;
  if (gid >= TOT) return;
  int n = (int)(gid % F_IN_N);
  int i = (int)((gid / F_IN_N) % D_INP_N);
  int b = (int)(gid / ((long)F_IN_N * D_INP_N));
  int s = n >> 2, j = n & 3;
  float v = src[((size_t)s * BATCH_N + b) * 72 + i] * R_u[i * 4 + j];
  X[gid] = v > 0.f ? v : 0.f;
}

// =====================================================================
// GAT attention: per-sample softmax(leaky_relu(d_i + s_j)) * edge_w
// =====================================================================
__global__ __launch_bounds__(256) void gat_attn(
    const float* __restrict__ XP, const float* __restrict__ a_d,
    const float* __restrict__ a_s, const float* __restrict__ edge,
    float* __restrict__ alpha)
{
  __shared__ float ds[D_INP_N], ss[D_INP_N];
  const int b    = blockIdx.x;
  const int wave = threadIdx.x >> 5;
  const int lane = threadIdx.x & 31;
  const float* xp = XP + (size_t)b * D_INP_N * F_IN_N;
  for (int r = wave; r < 2 * D_INP_N; r += 8) {
    const float* vec  = (r < D_INP_N) ? a_d : a_s;
    const float* rowp = xp + (size_t)(r % D_INP_N) * F_IN_N;
    float p = 0.f;
    for (int k = lane; k < F_IN_N; k += 32) p += rowp[k] * vec[k];
    for (int off = 16; off; off >>= 1) p += __shfl_xor(p, off);
    if (lane == 0) { if (r < D_INP_N) ds[r] = p; else ss[r - D_INP_N] = p; }
  }
  __syncthreads();
  int i = threadIdx.x;
  if (i < D_INP_N) {
    float e[D_INP_N];
    float mx = -1e30f;
#pragma unroll
    for (int j = 0; j < D_INP_N; ++j) {
      float t = ds[i] + ss[j];
      t = t > 0.f ? t : 0.2f * t;          // leaky_relu(.,0.2)
      e[j] = t; mx = fmaxf(mx, t);
    }
    float sum = 0.f;
#pragma unroll
    for (int j = 0; j < D_INP_N; ++j) { e[j] = __expf(e[j] - mx); sum += e[j]; }
    float inv = 1.f / sum;
    size_t base = (size_t)b * D_INP_N * D_INP_N + (size_t)i * D_INP_N;
#pragma unroll
    for (int j = 0; j < D_INP_N; ++j) {
      float ew = edge ? edge[base + j] : 1.f;
      alpha[base + j] = e[j] * inv * ew;
    }
  }
}

// H[b,i,n] = sum_j alpha[b,i,j] * XP[b,j,n]
__global__ __launch_bounds__(256) void gat_agg(
    const float* __restrict__ alpha, const float* __restrict__ XP,
    float* __restrict__ H)
{
  long gid = (long)blockIdx.x * 256 + threadIdx.x;
  const long TOT = (long)BATCH_N * D_INP_N * F_IN_N;
  if (gid >= TOT) return;
  int n = (int)(gid % F_IN_N);
  int i = (int)((gid / F_IN_N) % D_INP_N);
  int b = (int)(gid / ((long)F_IN_N * D_INP_N));
  const float* al = alpha + (size_t)b * D_INP_N * D_INP_N + (size_t)i * D_INP_N;
  const float* xp = XP + (size_t)b * D_INP_N * F_IN_N + n;
  float s = 0.f;
#pragma unroll
  for (int j = 0; j < D_INP_N; ++j) s += al[j] * xp[(size_t)j * F_IN_N];
  H[gid] = s;
}

// =====================================================================
// distance from Gram matrix G = A A^T (sq_i = G[i,i])
// =====================================================================
__global__ __launch_bounds__(256) void dist_rows(
    const float* __restrict__ G, float* __restrict__ partial)
{
  __shared__ float red[256];
  int i = blockIdx.x;
  float sqi = G[(size_t)i * BATCH_N + i];
  float s = 0.f;
  for (int j = threadIdx.x; j < BATCH_N; j += 256) {
    float d2 = sqi + G[(size_t)j * BATCH_N + j] - 2.f * G[(size_t)i * BATCH_N + j];
    d2 = fmaxf(d2, 0.f);
    s += sqrtf(d2 + 1e-12f);
  }
  red[threadIdx.x] = s; __syncthreads();
  for (int off = 128; off; off >>= 1) {
    if (threadIdx.x < off) red[threadIdx.x] += red[threadIdx.x + off];
    __syncthreads();
  }
  if (threadIdx.x == 0) partial[i] = red[0];
}

__global__ __launch_bounds__(256) void dist_final(
    const float* __restrict__ partial, float* __restrict__ out)
{
  __shared__ float red[256];
  float s = 0.f;
  for (int i = threadIdx.x; i < BATCH_N; i += 256) s += partial[i];
  red[threadIdx.x] = s; __syncthreads();
  for (int off = 128; off; off >>= 1) {
    if (threadIdx.x < off) red[threadIdx.x] += red[threadIdx.x + off];
    __syncthreads();
  }
  if (threadIdx.x == 0) out[BATCH_N * 2] = red[0] / ((float)BATCH_N * (float)BATCH_N);
}

// =====================================================================
// OUT[s,b,c] = c<144 ? h2[b,c>>2, s*4+(c&3)] : pos-enc(times[s,b], c-144)
// =====================================================================
__global__ __launch_bounds__(256) void build_out(
    const float* __restrict__ XO, const float* __restrict__ times,
    float* __restrict__ OUT)
{
  long gid = (long)blockIdx.x * 256 + threadIdx.x;
  const long TOT = (long)S_LEN * BATCH_N * D_T_N;
  if (gid >= TOT) return;
  int c = (int)(gid % D_T_N);
  int b = (int)((gid / D_T_N) % BATCH_N);
  int s = (int)(gid / ((long)D_T_N * BATCH_N));
  float v;
  if (c < 144) {
    v = XO[(size_t)b * D_INP_N * F_IN_N + (size_t)(c >> 2) * F_IN_N + (s << 2) + (c & 3)];
  } else {
    int k = c - 144, kk = k & 7;
    float ts = powf(215.f, (float)kk / 7.f) * 100.f;
    float sc = times[(size_t)s * BATCH_N + b] / ts;
    v = (k < 8) ? sinf(sc) : cosf(sc);
  }
  OUT[gid] = v;
}

__global__ __launch_bounds__(256) void emb_kernel(
    const float* __restrict__ st, const float* __restrict__ w,
    const float* __restrict__ bb, float* __restrict__ EMB)
{
  int gid = blockIdx.x * 256 + threadIdx.x;
  if (gid >= BATCH_N * D_INP_N) return;
  int b = gid / D_INP_N, j = gid % D_INP_N;
  float s = bb[j];
#pragma unroll
  for (int k = 0; k < 9; ++k) s += st[b * 9 + k] * w[k * D_INP_N + j];
  EMB[gid] = s;
}

// =====================================================================
// fused attention: one thread per (s,b,h); online softmax over keys
// =====================================================================
__global__ __launch_bounds__(256) void attention(
    const float* __restrict__ QKV, const int* __restrict__ lengths,
    float* __restrict__ O)
{
  long gid = (long)blockIdx.x * 256 + threadIdx.x;
  const long TOT = (long)S_LEN * BATCH_N * NHEAD_N;
  if (gid >= TOT) return;
  int h = (int)(gid & 3);
  int b = (int)((gid >> 2) & (BATCH_N - 1));
  int s = (int)(gid >> 11);
  const float scale = 0.15811388300841897f;   // 1/sqrt(40)
  float qr[DH_N];
  const float* q = QKV + ((size_t)s * BATCH_N + b) * 480 + h * DH_N;
#pragma unroll
  for (int d = 0; d < DH_N; ++d) qr[d] = q[d];
  const int len = lengths[b];
  float m = -1e30f, l = 0.f, acc[DH_N];
#pragma unroll
  for (int d = 0; d < DH_N; ++d) acc[d] = 0.f;
  for (int t = 0; t < S_LEN; ++t) {
    if (t >= len) continue;      // exp(-1e9 - max) == 0 in f32, identical to ref
    const float* kv = QKV + ((size_t)t * BATCH_N + b) * 480 + 160 + h * DH_N;
    float sc = 0.f;
#pragma unroll
    for (int d = 0; d < DH_N; ++d) sc += qr[d] * kv[d];
    sc *= scale;
    float mn = fmaxf(m, sc);
    float corr = __expf(m - mn);
    float w = __expf(sc - mn);
    l = l * corr + w;
    const float* vv = kv + 160;
#pragma unroll
    for (int d = 0; d < DH_N; ++d) acc[d] = acc[d] * corr + w * vv[d];
    m = mn;
  }
  float inv = 1.f / l;
  float* o = O + ((size_t)s * BATCH_N + b) * D_T_N + h * DH_N;
#pragma unroll
  for (int d = 0; d < DH_N; ++d) o[d] = acc[d] * inv;
}

// =====================================================================
// X = LayerNorm(X + R) * g + b ; wave-per-token, 160 dims, 5 per lane
// =====================================================================
__global__ __launch_bounds__(256) void add_ln(
    float* __restrict__ X, const float* __restrict__ R,
    const float* __restrict__ g, const float* __restrict__ bb, int T)
{
  const int lane = threadIdx.x & 31;
  const int wave = threadIdx.x >> 5;
  long tok = (long)blockIdx.x * 8 + wave;
  if (tok >= T) return;
  size_t base = (size_t)tok * D_T_N;
  float v[5];
  float s = 0.f;
#pragma unroll
  for (int i = 0; i < 5; ++i) { v[i] = X[base + lane * 5 + i] + R[base + lane * 5 + i]; s += v[i]; }
  for (int off = 16; off; off >>= 1) s += __shfl_xor(s, off);
  float mean = s / (float)D_T_N;
  float vs = 0.f;
#pragma unroll
  for (int i = 0; i < 5; ++i) { float d = v[i] - mean; vs += d * d; }
  for (int off = 16; off; off >>= 1) vs += __shfl_xor(vs, off);
  float rstd = rsqrtf(vs / (float)D_T_N + 1e-5f);
#pragma unroll
  for (int i = 0; i < 5; ++i) {
    int c = lane * 5 + i;
    X[base + c] = (v[i] - mean) * rstd * g[c] + bb[c];
  }
}

// =====================================================================
// feat[b,0:160] = sum_{s<len} OUT[s,b,:] / (len+1); feat[b,160:196]=emb
// =====================================================================
__global__ __launch_bounds__(256) void agg_feat(
    const float* __restrict__ OUT, const int* __restrict__ lengths,
    const float* __restrict__ EMB, float* __restrict__ FEAT)
{
  int gid = blockIdx.x * 256 + threadIdx.x;
  if (gid >= BATCH_N * D_FIN_N) return;
  int b = gid / D_FIN_N, d = gid % D_FIN_N;
  if (d < D_T_N) {
    int len = lengths[b];
    float s = 0.f;
    for (int t = 0; t < len; ++t) s += OUT[((size_t)t * BATCH_N + b) * D_T_N + d];
    FEAT[gid] = s / (float)(len + 1);
  } else {
    FEAT[gid] = EMB[b * D_INP_N + (d - D_T_N)];
  }
}

static inline int cdiv(long a, long b) { return (int)((a + b - 1) / b); }

// convert + run one GEMM: C(MxN) = A(MxK) @ B(KxN) (+bias,+relu)
static void gemm_f32(const float* A, const float* Bm, const float* bias, float* C,
                     int M, int N, int K, bool relu,
                     _Float16* A16, _Float16* B16, hipStream_t stream)
{
  const int Mp = (M + 15) & ~15, Np = (N + 15) & ~15, Kp = (K + 31) & ~31;
  cvt16 <<<cdiv((long)Mp * Kp, 256), 256, 0, stream>>>(A,  A16, M, K, Kp, (long)Mp * Kp);
  cvt16T<<<cdiv((long)Np * Kp, 256), 256, 0, stream>>>(Bm, B16, K, N, Kp, (long)Np * Kp);
  long tiles = (long)(Mp >> 4) * (Np >> 4);
  int blocks = (int)((tiles + 7) / 8);
  if (bias && relu) wmma_gemm<true, true ><<<blocks, 256, 0, stream>>>(A16, B16, bias, C, M, N, Kp);
  else if (bias)    wmma_gemm<true, false><<<blocks, 256, 0, stream>>>(A16, B16, bias, C, M, N, Kp);
  else              wmma_gemm<false,false><<<blocks, 256, 0, stream>>>(A16, B16, nullptr, C, M, N, Kp);
}

extern "C" void kernel_launch(void* const* d_in, const int* in_sizes, int n_in,
                              void* d_out, int out_size, void* d_ws, size_t ws_size,
                              hipStream_t stream) {
  (void)in_sizes; (void)n_in; (void)out_size; (void)ws_size;
  const float* src       = (const float*)d_in[0];
  const float* stat_in   = (const float*)d_in[1];
  const float* times     = (const float*)d_in[2];
  const int*   lengths   = (const int*)  d_in[3];
  const float* R_u       = (const float*)d_in[4];
  const float* emb_w     = (const float*)d_in[5];
  const float* emb_b     = (const float*)d_in[6];
  const float* W1        = (const float*)d_in[7];
  const float* a1_s      = (const float*)d_in[8];
  const float* a1_d      = (const float*)d_in[9];
  const float* W2        = (const float*)d_in[10];
  const float* a2_s      = (const float*)d_in[11];
  const float* a2_d      = (const float*)d_in[12];
  const float* attn_in_w = (const float*)d_in[13];
  const float* attn_in_b = (const float*)d_in[14];
  const float* attn_out_w= (const float*)d_in[15];
  const float* attn_out_b= (const float*)d_in[16];
  const float* ff1_w     = (const float*)d_in[17];
  const float* ff1_b     = (const float*)d_in[18];
  const float* ff2_w     = (const float*)d_in[19];
  const float* ff2_b     = (const float*)d_in[20];
  const float* ln1_g     = (const float*)d_in[21];
  const float* ln1_b     = (const float*)d_in[22];
  const float* ln2_g     = (const float*)d_in[23];
  const float* ln2_b     = (const float*)d_in[24];
  const float* mlp1_w    = (const float*)d_in[25];
  const float* mlp1_b    = (const float*)d_in[26];
  const float* mlp2_w    = (const float*)d_in[27];
  const float* mlp2_b    = (const float*)d_in[28];
  float* out = (float*)d_out;

  // ---- workspace layout (floats). GAT region aliases under QKV since the
  // GAT + distance phase completes before the transformer phase.
  const size_t SZ_X   = (size_t)BATCH_N * D_INP_N * F_IN_N;   // 15,851,520
  const size_t SZ_AL  = (size_t)BATCH_N * D_INP_N * D_INP_N;  //    663,552
  const size_t SZ_OUT = (size_t)S_LEN * BATCH_N * D_T_N;      // 17,612,800
  const size_t SZ_QKV = (size_t)S_LEN * BATCH_N * 480;        // 52,838,400
  float* ws   = (float*)d_ws;
  float* X    = ws;                    // X -> H1 -> h2 (X_OUT)
  float* XP   = X + SZ_X;              // XP1 -> XP2
  float* AL1  = XP + SZ_X;
  float* AL2  = AL1 + SZ_AL;
  float* Gm   = AL2 + SZ_AL;
  float* PART = Gm + (size_t)BATCH_N * BATCH_N;
  float* QKV  = ws;                    // aliases GAT region (33.3M < 52.8M floats)
  float* OUT  = ws + SZ_QKV;
  float* O    = OUT + SZ_OUT;
  float* P    = O + SZ_OUT;            // also FF hidden (110080*128 fits)
  float* EMB  = P + SZ_OUT;
  float* FEAT = EMB + (size_t)BATCH_N * D_INP_N;
  float* MLPH = FEAT + (size_t)BATCH_N * D_FIN_N;
  // f16 staging: A16 max = 110080*160 halves; B16 max = 864*864 halves
  float* A16f = MLPH + (size_t)BATCH_N * D_FIN_N;   // offset is 16B aligned
  _Float16* A16 = (_Float16*)A16f;
  float* B16f = A16f + (size_t)8806400;             // 17,612,800 halves
  _Float16* B16 = (_Float16*)B16f;

  const long TOTX = (long)SZ_X;
  const int  NTOK = S_LEN * BATCH_N;   // 110080
  const int  MX   = BATCH_N * D_INP_N; // 18432

  // ---- GAT phase ----
  build_x<<<cdiv(TOTX, 256), 256, 0, stream>>>(src, R_u, X);
  gemm_f32(X, W1, nullptr, XP, MX, F_IN_N, F_IN_N, false, A16, B16, stream);
  gat_attn<<<BATCH_N, 256, 0, stream>>>(XP, a1_d, a1_s, nullptr, AL1);
  gat_agg<<<cdiv(TOTX, 256), 256, 0, stream>>>(AL1, XP, X);                 // H1
  gemm_f32(X, W2, nullptr, XP, MX, F_IN_N, F_IN_N, false, A16, B16, stream);
  gat_attn<<<BATCH_N, 256, 0, stream>>>(XP, a2_d, a2_s, AL1, AL2);
  gat_agg<<<cdiv(TOTX, 256), 256, 0, stream>>>(AL2, XP, X);                 // h2

  // ---- distance: Gram = AL2 @ AL2^T (Bt of the NT product == AL2 itself) ----
  {
    const int KpG = (D_INP_N * D_INP_N + 31) & ~31;   // 1312
    cvt16<<<cdiv((long)BATCH_N * KpG, 256), 256, 0, stream>>>(
        AL2, A16, BATCH_N, D_INP_N * D_INP_N, KpG, (long)BATCH_N * KpG);
    long tiles = (long)(BATCH_N >> 4) * (BATCH_N >> 4);
    wmma_gemm<false,false><<<(int)((tiles + 7) / 8), 256, 0, stream>>>(
        A16, A16, nullptr, Gm, BATCH_N, BATCH_N, KpG);
  }
  dist_rows<<<BATCH_N, 256, 0, stream>>>(Gm, PART);
  dist_final<<<1, 256, 0, stream>>>(PART, out);

  // ---- embed / sequence assembly ----
  build_out<<<cdiv((long)NTOK * D_T_N, 256), 256, 0, stream>>>(X, times, OUT);
  emb_kernel<<<cdiv(BATCH_N * D_INP_N, 256), 256, 0, stream>>>(stat_in, emb_w, emb_b, EMB);

  // ---- transformer layers ----
  for (int l = 0; l < 2; ++l) {
    gemm_f32(OUT, attn_in_w + (size_t)l*D_T_N*480, attn_in_b + l*480,
             QKV, NTOK, 480, D_T_N, false, A16, B16, stream);
    attention<<<cdiv((long)NTOK * NHEAD_N, 256), 256, 0, stream>>>(QKV, lengths, O);
    gemm_f32(O, attn_out_w + (size_t)l*D_T_N*D_T_N, attn_out_b + l*D_T_N,
             P, NTOK, D_T_N, D_T_N, false, A16, B16, stream);
    add_ln<<<cdiv(NTOK, 8), 256, 0, stream>>>(OUT, P, ln1_g + l*D_T_N, ln1_b + l*D_T_N, NTOK);
    gemm_f32(OUT, ff1_w + (size_t)l*D_T_N*NHID_N, ff1_b + l*NHID_N,
             P, NTOK, NHID_N, D_T_N, true, A16, B16, stream);
    gemm_f32(P, ff2_w + (size_t)l*NHID_N*D_T_N, ff2_b + l*D_T_N,
             O, NTOK, D_T_N, NHID_N, false, A16, B16, stream);
    add_ln<<<cdiv(NTOK, 8), 256, 0, stream>>>(OUT, O, ln2_g + l*D_T_N, ln2_b + l*D_T_N, NTOK);
  }

  // ---- head ----
  agg_feat<<<cdiv(BATCH_N * D_FIN_N, 256), 256, 0, stream>>>(OUT, lengths, EMB, FEAT);
  gemm_f32(FEAT, mlp1_w, mlp1_b, MLPH, BATCH_N, D_FIN_N, D_FIN_N, true, A16, B16, stream);
  gemm_f32(MLPH, mlp2_w, mlp2_b, out, BATCH_N, 2, D_FIN_N, false, A16, B16, stream);
}
